// DCN1_80719615361475
// MI455X (gfx1250) — matrix operand order repaired and checked by reading, not compiled, not run on hardware
//
#include <hip/hip_runtime.h>
#include <hip/hip_bf16.h>
#include <math.h>

// ---------------- problem constants (match reference) ----------------
#define B_ROWS  16384
#define DENSE   64
#define N_OH    8
#define N_MH    4
#define VOCAB   100000
#define EMB     32
#define HIST    20
#define IN_DIM  448          // 64 + 8*32 + 4*32
#define NCHUNK  14           // 448 / 32
#define H1      1024
#define H2      512
#define H3      256
#define CROSS_ORDER 4

typedef _Float16 v16h __attribute__((ext_vector_type(16)));
typedef _Float16 v8h  __attribute__((ext_vector_type(8)));
typedef float    v8f  __attribute__((ext_vector_type(8)));

union HV16 { v16h v; v8h h[2]; };

// Byte-offset fragment loads: uniform base (SGPR pair) + 32-bit per-lane
// byte offset -> saddr+voffset global_load_b128 with immediate offsets.
// A-fragment halves are +0 / +16 bytes; B-fragment halves +0 / +16 bytes.
__device__ __forceinline__ void load_fragA(HV16& f, const char* base,
                                           unsigned byteOff) {
  f.h[0] = *(const v8h*)(base + byteOff);
  f.h[1] = *(const v8h*)(base + byteOff + 16);
}
__device__ __forceinline__ void load_fragB(HV16& f, const char* base,
                                           unsigned byteOff) {
  f.h[0] = *(const v8h*)(base + byteOff);
  f.h[1] = *(const v8h*)(base + byteOff + 16);
}

// ---------------------------------------------------------------------
// W (K x N, f32, row-major) -> Wt (N x K, f16, row-major). One-shot, tiny.
// ---------------------------------------------------------------------
__global__ void k_w_to_f16_t(const float* __restrict__ W,
                             _Float16* __restrict__ Wt,
                             int K, int N) {
  int idx = blockIdx.x * blockDim.x + threadIdx.x;
  if (idx >= K * N) return;
  int k = idx / N;
  int n = idx - k * N;
  Wt[(size_t)n * K + k] = (_Float16)W[idx];
}

// ---------------------------------------------------------------------
// Fused: build x0 (dense || one-hot gathers || multi-hot sum-pool),
// run 4 cross layers entirely in registers (wave32 per row),
// emit xi (f32, for final head) and x0 (f16, GEMM1 A-matrix).
// Chunk layout: element e = c*32 + lane  (c = 0..13)
// ---------------------------------------------------------------------
__global__ void k_embed_cross(const float* __restrict__ dense,
                              const int*   __restrict__ oh_idx,
                              const int*   __restrict__ mh_idx,
                              const float* __restrict__ oh_emb,
                              const float* __restrict__ mh_emb,
                              const float* __restrict__ cross_w,
                              const float* __restrict__ cross_b,
                              float*       __restrict__ xi_out,
                              _Float16*    __restrict__ x0_out) {
  const int lane = threadIdx.x & 31;
  const int wave = threadIdx.x >> 5;
  const int row  = blockIdx.x * 8 + wave;

  // preload all indices first so the gathers can overlap
  int ohi[N_OH];
#pragma unroll
  for (int f = 0; f < N_OH; ++f) ohi[f] = oh_idx[row * N_OH + f];

  float x0c[NCHUNK];
  x0c[0] = dense[(size_t)row * DENSE + lane];
  x0c[1] = dense[(size_t)row * DENSE + 32 + lane];
#pragma unroll
  for (int f = 0; f < N_OH; ++f)
    x0c[2 + f] = oh_emb[((size_t)f * VOCAB + ohi[f]) * EMB + lane];

#pragma unroll
  for (int f = 0; f < N_MH; ++f) {
    int idx[HIST];
#pragma unroll
    for (int j = 0; j < HIST; ++j)
      idx[j] = mh_idx[((size_t)row * N_MH + f) * HIST + j];
    float s = 0.f;
#pragma unroll
    for (int j = 0; j < HIST; ++j)
      s += mh_emb[((size_t)f * VOCAB + idx[j]) * EMB + lane];
    x0c[10 + f] = s;
  }

  // CrossNet: xi = x0 * dot(xi, w_l) + b_l + xi
  float xic[NCHUNK];
#pragma unroll
  for (int c = 0; c < NCHUNK; ++c) xic[c] = x0c[c];

#pragma unroll
  for (int l = 0; l < CROSS_ORDER; ++l) {
    float p = 0.f;
#pragma unroll
    for (int c = 0; c < NCHUNK; ++c)
      p += xic[c] * cross_w[l * IN_DIM + c * 32 + lane];
#pragma unroll
    for (int off = 16; off > 0; off >>= 1)
      p += __shfl_xor(p, off, 32);
#pragma unroll
    for (int c = 0; c < NCHUNK; ++c)
      xic[c] = x0c[c] * p + cross_b[l * IN_DIM + c * 32 + lane] + xic[c];
  }

#pragma unroll
  for (int c = 0; c < NCHUNK; ++c) {
    int e = c * 32 + lane;
    xi_out[(size_t)row * IN_DIM + e] = xic[c];
    x0_out[(size_t)row * IN_DIM + e] = (_Float16)x0c[c];
  }
}

// ---------------------------------------------------------------------
// WMMA GEMM: Out(M x N, f16) = relu(A(M x K, f16) @ Wt^T + bias)
//   Wt stored N x K (K-major) so B-fragments are contiguous.
// Wave -> 32x64 strip: 2 M-tiles x 4 N-tiles = 8 accumulators.
// Block (8 waves = 2Mw x 4Nw) -> 64 x 256 tile.
// Addressing: uniform byte cursors advanced with scalar adds + fixed 32-bit
// per-lane byte offsets -> saddr+voffset loads, no per-lane address math in
// the K-loop.  K-loop is ping-pong software-pipelined; requires K % 64 == 0
// (448, 1024, 512 all qualify).
// ---------------------------------------------------------------------
__global__ void __launch_bounds__(256, 2)
k_gemm_relu_f16(const _Float16* __restrict__ A,
                const _Float16* __restrict__ Bt,
                const float*    __restrict__ bias,
                _Float16*       __restrict__ Out,
                int N, int K) {
  const int lane  = threadIdx.x & 31;
  const int wave  = threadIdx.x >> 5;
  const int wm    = wave & 1;          // 2 waves along M
  const int wn    = wave >> 1;         // 4 waves along N
  const int row0  = blockIdx.y * 64 + wm * 32;   // wave owns rows row0..row0+31
  const int colw  = blockIdx.x * 256 + wn * 64;  // wave owns cols colw..colw+63
  const int mn    = lane & 15;         // M within A-frag / N within B-frag
  const int halfk = lane >> 4;

  // fixed per-lane BYTE offsets (all < 2^31)
  const unsigned offA0 = ((unsigned)(row0 + mn) * (unsigned)K + halfk * 16) * 2u;
  const unsigned offA1 = offA0 + 32u * (unsigned)K;          // +16 rows
  unsigned offB[4];
#pragma unroll
  for (int t = 0; t < 4; ++t)
    offB[t] = ((unsigned)(colw + t * 16 + mn) * (unsigned)K + halfk * 16) * 2u;

  // uniform byte cursors, advanced by scalar adds
  const char* pA = (const char*)A;
  const char* pB = (const char*)Bt;

  v8f acc[2][4] = {};
  HV16 a0[2], a1[2], b0[4], b1[4];

  // prologue: K-step 0 into buffer 0
  load_fragA(a0[0], pA, offA0);
  load_fragA(a0[1], pA, offA1);
#pragma unroll
  for (int t = 0; t < 4; ++t) load_fragB(b0[t], pB, offB[t]);

  const int pairsMinus1 = (K >> 6) - 1;   // K/64 - 1 double-steps in the loop
  for (int it = 0; it < pairsMinus1; ++it) {
    // load step +32 (buffer 1), consume buffer 0
    load_fragA(a1[0], pA + 64, offA0);
    load_fragA(a1[1], pA + 64, offA1);
#pragma unroll
    for (int t = 0; t < 4; ++t) load_fragB(b1[t], pB + 64, offB[t]);
#pragma unroll
    for (int t = 0; t < 4; ++t) {
      acc[0][t] = __builtin_amdgcn_wmma_f32_16x16x32_f16(
          false, a0[0].v, false, b0[t].v, (short)0, acc[0][t], false, false);
      acc[1][t] = __builtin_amdgcn_wmma_f32_16x16x32_f16(
          false, a0[1].v, false, b0[t].v, (short)0, acc[1][t], false, false);
    }
    // load step +64 (buffer 0), consume buffer 1
    load_fragA(a0[0], pA + 128, offA0);
    load_fragA(a0[1], pA + 128, offA1);
#pragma unroll
    for (int t = 0; t < 4; ++t) load_fragB(b0[t], pB + 128, offB[t]);
#pragma unroll
    for (int t = 0; t < 4; ++t) {
      acc[0][t] = __builtin_amdgcn_wmma_f32_16x16x32_f16(
          false, a1[0].v, false, b1[t].v, (short)0, acc[0][t], false, false);
      acc[1][t] = __builtin_amdgcn_wmma_f32_16x16x32_f16(
          false, a1[1].v, false, b1[t].v, (short)0, acc[1][t], false, false);
    }
    pA += 128;   // 64 f16 elements
    pB += 128;
  }

  // epilogue: last two K-steps
  load_fragA(a1[0], pA + 64, offA0);
  load_fragA(a1[1], pA + 64, offA1);
#pragma unroll
  for (int t = 0; t < 4; ++t) load_fragB(b1[t], pB + 64, offB[t]);
#pragma unroll
  for (int t = 0; t < 4; ++t) {
    acc[0][t] = __builtin_amdgcn_wmma_f32_16x16x32_f16(
        false, a0[0].v, false, b0[t].v, (short)0, acc[0][t], false, false);
    acc[1][t] = __builtin_amdgcn_wmma_f32_16x16x32_f16(
        false, a0[1].v, false, b0[t].v, (short)0, acc[1][t], false, false);
  }
#pragma unroll
  for (int t = 0; t < 4; ++t) {
    acc[0][t] = __builtin_amdgcn_wmma_f32_16x16x32_f16(
        false, a1[0].v, false, b1[t].v, (short)0, acc[0][t], false, false);
    acc[1][t] = __builtin_amdgcn_wmma_f32_16x16x32_f16(
        false, a1[1].v, false, b1[t].v, (short)0, acc[1][t], false, false);
  }

  // store: C layout lane -> row = v + 8*(lane>>4), col = lane&15
#pragma unroll
  for (int mt = 0; mt < 2; ++mt) {
#pragma unroll
    for (int t = 0; t < 4; ++t) {
      const int col = colw + t * 16 + mn;
      const float bv = bias[col];
#pragma unroll
      for (int v = 0; v < 8; ++v) {
        const int rm = v + 8 * halfk;
        float val = acc[mt][t][v] + bv;
        val = val > 0.f ? val : 0.f;
        Out[(size_t)(row0 + mt * 16 + rm) * N + col] = (_Float16)val;
      }
    }
  }
}

// ---------------------------------------------------------------------
// Head: out = sigmoid( dot(xi, lin_w[0:448]) + dot(h3, lin_w[448:704]) + lin_b )
// wave32 per row.
// ---------------------------------------------------------------------
__global__ void k_final(const float*    __restrict__ xi,
                        const _Float16* __restrict__ h3,
                        const float*    __restrict__ lin_w,
                        const float*    __restrict__ lin_b,
                        float*          __restrict__ out) {
  const int lane = threadIdx.x & 31;
  const int wave = threadIdx.x >> 5;
  const int row  = blockIdx.x * 8 + wave;

  float p = 0.f;
#pragma unroll
  for (int c = 0; c < NCHUNK; ++c) {
    int e = c * 32 + lane;
    p += xi[(size_t)row * IN_DIM + e] * lin_w[e];
  }
#pragma unroll
  for (int c = 0; c < H3 / 32; ++c) {
    int e = c * 32 + lane;
    p += (float)h3[(size_t)row * H3 + e] * lin_w[IN_DIM + e];
  }
#pragma unroll
  for (int off = 16; off > 0; off >>= 1)
    p += __shfl_xor(p, off, 32);

  if (lane == 0) {
    float z = p + lin_b[0];
    out[row] = 1.f / (1.f + __expf(-z));
  }
}

// ---------------------------------------------------------------------
extern "C" void kernel_launch(void* const* d_in, const int* in_sizes, int n_in,
                              void* d_out, int out_size, void* d_ws, size_t ws_size,
                              hipStream_t stream) {
  (void)in_sizes; (void)n_in; (void)out_size; (void)ws_size;

  const float* dense  = (const float*)d_in[0];
  const int*   oh_idx = (const int*)  d_in[1];
  const int*   mh_idx = (const int*)  d_in[2];
  const float* oh_emb = (const float*)d_in[3];
  const float* mh_emb = (const float*)d_in[4];
  const float* cw     = (const float*)d_in[5];
  const float* cb     = (const float*)d_in[6];
  const float* W1     = (const float*)d_in[7];
  const float* b1     = (const float*)d_in[8];
  const float* W2     = (const float*)d_in[9];
  const float* b2     = (const float*)d_in[10];
  const float* W3     = (const float*)d_in[11];
  const float* b3     = (const float*)d_in[12];
  const float* lin_w  = (const float*)d_in[13];
  const float* lin_b  = (const float*)d_in[14];
  float* out = (float*)d_out;

  // workspace carve-up (256B aligned slabs)
  char* ws = (char*)d_ws;
  size_t off = 0;
  auto carve = [&](size_t bytes) -> void* {
    void* p = ws + off;
    off += (bytes + 255) & ~(size_t)255;
    return p;
  };
  _Float16* x0h  = (_Float16*)carve((size_t)B_ROWS * IN_DIM * 2);
  float*    xi   = (float*)   carve((size_t)B_ROWS * IN_DIM * 4);
  _Float16* h1h  = (_Float16*)carve((size_t)B_ROWS * H1 * 2);
  _Float16* h2h  = (_Float16*)carve((size_t)B_ROWS * H2 * 2);
  _Float16* h3h  = (_Float16*)carve((size_t)B_ROWS * H3 * 2);
  _Float16* W1t  = (_Float16*)carve((size_t)IN_DIM * H1 * 2);
  _Float16* W2t  = (_Float16*)carve((size_t)H1 * H2 * 2);
  _Float16* W3t  = (_Float16*)carve((size_t)H2 * H3 * 2);

  // 1) weight transpose/convert (tiny one-shots)
  k_w_to_f16_t<<<(IN_DIM * H1 + 255) / 256, 256, 0, stream>>>(W1, W1t, IN_DIM, H1);
  k_w_to_f16_t<<<(H1 * H2 + 255) / 256, 256, 0, stream>>>(W2, W2t, H1, H2);
  k_w_to_f16_t<<<(H2 * H3 + 255) / 256, 256, 0, stream>>>(W3, W3t, H2, H3);

  // 2) embeddings + cross net (wave per row, 8 rows/block)
  k_embed_cross<<<B_ROWS / 8, 256, 0, stream>>>(
      dense, oh_idx, mh_idx, oh_emb, mh_emb, cw, cb, xi, x0h);

  // 3) deep net: three WMMA GEMMs with fused bias+relu, f16 activations
  k_gemm_relu_f16<<<dim3(H1 / 256, B_ROWS / 64), 256, 0, stream>>>(
      x0h, W1t, b1, h1h, H1, IN_DIM);
  k_gemm_relu_f16<<<dim3(H2 / 256, B_ROWS / 64), 256, 0, stream>>>(
      h1h, W2t, b2, h2h, H2, H1);
  k_gemm_relu_f16<<<dim3(H3 / 256, B_ROWS / 64), 256, 0, stream>>>(
      h2h, W3t, b3, h3h, H3, H2);

  // 4) head: concat-dot + sigmoid
  k_final<<<B_ROWS / 8, 256, 0, stream>>>(xi, h3h, lin_w, lin_b, out);
}
